// GraphConvLayer_87771951661495
// MI455X (gfx1250) — matrix operand order repaired and checked by reading, not compiled, not run on hardware
//
#include <hip/hip_runtime.h>

// ---- CDNA5 vector types ----
typedef __attribute__((ext_vector_type(16))) __bf16          v16bf;
typedef __attribute__((ext_vector_type(8)))  float           v8f;
typedef __attribute__((ext_vector_type(16))) unsigned short  v16u;
typedef __attribute__((ext_vector_type(8)))  unsigned short  v8u;
typedef __attribute__((ext_vector_type(4)))  float           v4f;
typedef __attribute__((ext_vector_type(2)))  float           v2f;

#define B_  32
#define N_  1024
#define F_  128

// f32 -> bf16 round-to-nearest-even (bit trick; NaN edge cases ignored)
__device__ __forceinline__ unsigned short f2bf(float f) {
    unsigned u = __builtin_bit_cast(unsigned, f);
    u += 0x7FFFu + ((u >> 16) & 1u);
    return (unsigned short)(u >> 16);
}

// ---------------------------------------------------------------------------
// Kernel 0: weight [Fin,Fout] f32  ->  weightT [Fout,Fin] bf16 (32 KB)
// ---------------------------------------------------------------------------
__global__ void k_wt(const float* __restrict__ w, unsigned short* __restrict__ wT) {
    int i  = blockIdx.x * blockDim.x + threadIdx.x;   // 0..16383
    int o  = i & (F_ - 1);                            // out col (contiguous reads)
    int in = i >> 7;                                  // in row
    wT[o * F_ + in] = f2bf(w[in * F_ + o]);
}

// ---------------------------------------------------------------------------
// Kernel 1: d[b,n] = rsqrt( sum_m adj[b,n,m] + 1 )   (rowsum >= 1, never inf)
// one wave32 per row, vectorized float4 loads, shuffle reduction
// ---------------------------------------------------------------------------
__global__ void k_deg(const float* __restrict__ adj, float* __restrict__ d) {
    int wave = threadIdx.x >> 5;
    int lane = threadIdx.x & 31;
    int row  = blockIdx.x * 8 + wave;                 // 0..B_*N_-1
    const v4f* p4 = (const v4f*)(adj + (size_t)row * N_);
    float s = 0.f;
    for (int j = lane; j < N_ / 4; j += 32) {
        v4f v = p4[j];
        s += v.x + v.y + v.z + v.w;
    }
    for (int off = 16; off > 0; off >>= 1) s += __shfl_xor(s, off, 32);
    if (lane == 0) d[row] = rsqrtf(s + 1.0f);
}

// ---------------------------------------------------------------------------
// Kernel 2: support = x @ W  (bf16 WMMA, f32 acc), stored transposed bf16:
//           sT[b][col][n].  16-row block per WG, 8 waves = 8 col tiles.
// LDS holds the 16x128 x-tile pre-swizzled into A-fragment order (4 KB).
// A-frag order: lane L (hi=L>>4, M=L&15), element e -> K = 16*hi + e
// ---------------------------------------------------------------------------
__global__ void k_support(const float* __restrict__ x,
                          const unsigned short* __restrict__ wT,
                          unsigned short* __restrict__ sT) {
    __shared__ __attribute__((aligned(64))) unsigned short lds[4 * 512];
    int t = threadIdx.x;
    int rowbase = blockIdx.x * 16;                    // global m base (B_*N_/16 blocks)

    // producer: 8 consecutive k per thread -> 8 contiguous frag elements
    {
        int M  = t >> 4;
        int k0 = (t & 15) * 8;
        const float* xp = x + (size_t)(rowbase + M) * F_ + k0;
        v4f a  = *(const v4f*)xp;
        v4f b2 = *(const v4f*)(xp + 4);
        v8u pack;
        pack[0] = f2bf(a.x);  pack[1] = f2bf(a.y);  pack[2] = f2bf(a.z);  pack[3] = f2bf(a.w);
        pack[4] = f2bf(b2.x); pack[5] = f2bf(b2.y); pack[6] = f2bf(b2.z); pack[7] = f2bf(b2.w);
        int kstep = k0 >> 5;
        int kk    = k0 & 31;                          // 0,8,16,24 -> constant hi, 8 contiguous e
        int hi    = kk >> 4;
        int e0    = kk & 15;
        *(v8u*)&lds[kstep * 512 + (M + 16 * hi) * 16 + e0] = pack;
    }
    __syncthreads();

    int wave = t >> 5, lane = t & 31;
    int col   = wave * 16 + (lane & 15);
    int khalf = 16 * (lane >> 4);
    v8f c = {};
#pragma unroll
    for (int ks = 0; ks < 4; ++ks) {
        v16bf a = __builtin_bit_cast(v16bf, *(const v16u*)&lds[ks * 512 + lane * 16]);
        v16bf b = __builtin_bit_cast(v16bf, *(const v16u*)&wT[col * F_ + ks * 32 + khalf]);
        c = __builtin_amdgcn_wmma_f32_16x16x32_bf16(false, a, false, b,
                                                    (short)0, c, false, false);
    }
    // D layout: element r -> row M = r + 8*(lane>>4); store transposed bf16
    int bb = rowbase >> 10;
    int n0 = (rowbase & (N_ - 1)) + 8 * (lane >> 4);
    v8u o;
#pragma unroll
    for (int r = 0; r < 8; ++r) o[r] = f2bf(c[r]);
    *(v8u*)&sT[((size_t)bb * F_ + col) * N_ + n0] = o;
}

// ---------------------------------------------------------------------------
// Kernel 3: out[b] = (D^-1/2 (A+I) D^-1/2) @ support + bias
// WG = (batch, 16 output rows); 8 waves own the 8 Fout col-tiles.
// 32 K-steps: 256 threads normalize a 16x32 adj tile straight into
// A-fragment-order LDS (double buffered, 1 barrier/step); each wave does one
// v_wmma_f32_16x16x32_bf16 per step; B-frag = 32 contiguous bytes of sT.
// ---------------------------------------------------------------------------
__global__ void k_gemm(const float* __restrict__ adj,
                       const unsigned short* __restrict__ sT,
                       const float* __restrict__ d,
                       const float* __restrict__ bias,
                       float* __restrict__ out) {
    __shared__ __attribute__((aligned(64))) unsigned short lds[2 * 512];
    int t  = threadIdx.x;
    int bb = blockIdx.y;
    int rowbase = blockIdx.x * 16;

    // producer coordinates: thread t -> (row r, col pair cpair..cpair+1)
    int r     = t >> 4;
    int cpair = (t & 15) * 2;
    const float* adjrow = adj + ((size_t)bb * N_ + rowbase + r) * N_;
    const float* dB     = d + bb * N_;
    float drow = dB[rowbase + r];
    int   grow = rowbase + r;
    int hi_p = cpair >> 4;
    int e_p  = cpair & 15;                           // even -> u32-packable pair
    unsigned* ldsw = (unsigned*)lds;
    int widx = (((r + 16 * hi_p) * 16 + e_p) >> 1);  // u32 index inside one buffer

    // consumer coordinates
    int wave = t >> 5, lane = t & 31;
    int col  = wave * 16 + (lane & 15);
    const unsigned short* sTp =
        sT + ((size_t)bb * F_ + col) * N_ + 16 * (lane >> 4);

    v8f acc = {};
    v2f aval = *(const v2f*)(adjrow + cpair);        // prefetch step 0
    v2f dcol = *(const v2f*)(dB + cpair);

#pragma unroll 2
    for (int step = 0; step < 32; ++step) {
        int kbase = step * 32;
        int p = step & 1;
        // normalize: (adj + I) * d[row] * d[k], convert to bf16, frag-order LDS
        float v0 = aval.x + ((grow == kbase + cpair)     ? 1.0f : 0.0f);
        float v1 = aval.y + ((grow == kbase + cpair + 1) ? 1.0f : 0.0f);
        v0 *= drow * dcol.x;
        v1 *= drow * dcol.y;
        ldsw[p * 256 + widx] =
            (unsigned)f2bf(v0) | ((unsigned)f2bf(v1) << 16);
        __syncthreads();                              // implies ds-wait; guards buffer reuse
        if (step < 31) {                              // prefetch next tile during WMMA
            aval = *(const v2f*)(adjrow + kbase + 32 + cpair);
            dcol = *(const v2f*)(dB + kbase + 32 + cpair);
        }
        v16bf a = __builtin_bit_cast(v16bf, *(const v16u*)&lds[p * 512 + lane * 16]);
        v16bf b = __builtin_bit_cast(v16bf, *(const v16u*)&sTp[kbase]);
        acc = __builtin_amdgcn_wmma_f32_16x16x32_bf16(false, a, false, b,
                                                      (short)0, acc, false, false);
    }

    // epilogue: D element q -> row rowbase + 8*(lane>>4) + q, fixed col
    float bv = bias[col];
    int nout = rowbase + 8 * (lane >> 4);
    float* op = out + ((size_t)bb * N_ + nout) * F_ + col;
#pragma unroll
    for (int q = 0; q < 8; ++q) op[q * F_] = acc[q] + bv;
}

// ---------------------------------------------------------------------------
extern "C" void kernel_launch(void* const* d_in, const int* in_sizes, int n_in,
                              void* d_out, int out_size, void* d_ws, size_t ws_size,
                              hipStream_t stream) {
    (void)in_sizes; (void)n_in; (void)out_size; (void)ws_size;
    const float* x    = (const float*)d_in[0];   // [B,N,Fin]
    const float* adj  = (const float*)d_in[1];   // [B,N,N]
    const float* w    = (const float*)d_in[2];   // [Fin,Fout]
    const float* bias = (const float*)d_in[3];   // [Fout]
    float* out = (float*)d_out;

    char* ws = (char*)d_ws;
    float*          dvec = (float*)ws;                                  // 128 KB
    unsigned short* wT   = (unsigned short*)(ws + 131072);              // 32 KB
    unsigned short* sT   = (unsigned short*)(ws + 131072 + 32768);      // 8 MB

    k_wt     <<<64,   256, 0, stream>>>(w, wT);
    k_deg    <<<4096, 256, 0, stream>>>(adj, dvec);   // warms adj into L2 (128MB < 192MB)
    k_support<<<2048, 256, 0, stream>>>(x, wT, sT);
    dim3 g3(N_ / 16, B_);
    k_gemm   <<<g3,   256, 0, stream>>>(adj, sT, dvec, bias, out);
}
